// MambaAutoencoder_41755672052460
// MI455X (gfx1250) — matrix operand order repaired and checked by reading, not compiled, not run on hardware
//
#include <hip/hip_runtime.h>
#include <hip/hip_bf16.h>
#include <cstdint>

// ---------------------------------------------------------------- constants
#define BB    4
#define LL    1024
#define OBS_  128
#define DM_   1024
#define DI_   2048
#define DS_   16
#define DC_   4
#define DTR_  64
#define LAT_  16
#define MROWS (BB*LL)      // 4096 token rows

typedef __bf16 bf16_t;
typedef __attribute__((ext_vector_type(16))) __bf16 bf16x16;
typedef __attribute__((ext_vector_type(8)))  __bf16 bf16x8;
typedef __attribute__((ext_vector_type(8)))  float  f32x8;
typedef __attribute__((ext_vector_type(4)))  unsigned int u32x4;
typedef __attribute__((ext_vector_type(8)))  int i32x8;
typedef __attribute__((ext_vector_type(4)))  int i32x4;

// ------------------------------------------------ WMMA fragment loaders
// A matrix 16x32 bf16 (M x K), row-major source. Per ISA layout:
// lane l: m = l&15, kb = 8*(l>>4); holds K in [kb,kb+8) and [kb+16,kb+24).
__device__ __forceinline__ bf16x16 load_a_frag(const bf16_t* __restrict__ base,
                                               int ld, int row0, int k0, int lane) {
  const int r  = row0 + (lane & 15);
  const int kb = (lane >> 4) << 3;
  const bf16_t* p = base + (size_t)r * ld + k0 + kb;
  bf16x8 lo = *(const bf16x8*)p;
  bf16x8 hi = *(const bf16x8*)(p + 16);
  bf16x16 f;
#pragma unroll
  for (int i = 0; i < 8; ++i) { f[i] = lo[i]; f[i + 8] = hi[i]; }
  return f;
}

// B matrix 32x16 bf16 (K x N), sourced from W stored (N x K) row-major.
// Contiguous k order (not the ISA table order): lane l: n = l&15, holds K in [16*(l>>4), +16).
__device__ __forceinline__ bf16x16 load_b_frag(const bf16_t* __restrict__ base,
                                               int ld, int row0, int k0, int lane) {
  const int r  = row0 + (lane & 15);
  const int kb = (lane >> 4) << 4;
  const bf16_t* p = base + (size_t)r * ld + k0 + kb;
  bf16x8 lo = *(const bf16x8*)p;
  bf16x8 hi = *(const bf16x8*)(p + 8);
  bf16x16 f;
#pragma unroll
  for (int i = 0; i < 8; ++i) { f[i] = lo[i]; f[i + 8] = hi[i]; }
  return f;
}

// B fragment out of an LDS-staged 32-wide row-major tile (row = n, 32 k elems).
__device__ __forceinline__ bf16x16 load_b_frag_lds(const bf16_t* tile,
                                                   int row0, int lane) {
  const int r  = row0 + (lane & 15);
  const int kb = (lane >> 4) << 3;
  const bf16_t* p = tile + r * 32 + kb;
  bf16x8 lo = *(const bf16x8*)p;
  bf16x8 hi = *(const bf16x8*)(p + 16);
  bf16x16 f;
#pragma unroll
  for (int i = 0; i < 8; ++i) { f[i] = lo[i]; f[i + 8] = hi[i]; }
  return f;
}

__device__ __forceinline__ float silu_f(float v) {
  return v * (1.0f / (1.0f + __expf(-v)));
}

// ------------------------------------------------ Tensor Data Mover issue
// 2-D tile load: Memory(row-major, stride_elems) -> LDS (compact rows).
// D# packing per CDNA5 ISA (group0 128b, group1 256b).
__device__ __forceinline__ void tdm_load_2d(uint64_t gaddr, unsigned int lds_off,
                                            unsigned int tdim0, unsigned int tdim1,
                                            unsigned int tile0, unsigned int tile1,
                                            uint64_t stride_elems) {
  u32x4 g0;
  g0[0] = 1u;                                             // count=1 (valid user D#)
  g0[1] = lds_off;                                        // lds_addr (bytes)
  g0[2] = (unsigned int)gaddr;                            // global_addr[31:0]
  g0[3] = ((unsigned int)(gaddr >> 32) & 0x01FFFFFFu)     // global_addr[56:32]
          | 0x80000000u;                                  // type=2 ("image")
  i32x8 g1;
  g1[0] = (int)(1u << 16);                                // data_size=1 -> 2 bytes
  g1[1] = (int)((tdim0 & 0xFFFFu) << 16);                 // tensor_dim0[15:0]
  g1[2] = (int)((tdim0 >> 16) | ((tdim1 & 0xFFFFu) << 16));
  g1[3] = (int)((tdim1 >> 16) | (tile0 << 16));           // tile_dim0
  g1[4] = (int)(tile1 & 0xFFFFu);                         // tile_dim1 (tile_dim2=0)
  g1[5] = (int)(unsigned int)stride_elems;                // tensor_dim0_stride lo
  g1[6] = (int)((unsigned int)(stride_elems >> 32) & 0xFFFFu);
  g1[7] = 0;
  i32x4 z4 = {0, 0, 0, 0};
#if __clang_major__ >= 23
  i32x8 z8 = {0, 0, 0, 0, 0, 0, 0, 0};
  __builtin_amdgcn_tensor_load_to_lds(g0, g1, z4, z4, z8, 0);
#else
  __builtin_amdgcn_tensor_load_to_lds(g0, g1, z4, z4, 0);
#endif
}

// ------------------------------------------------ generic WMMA GEMM
// C(MxN) = A(MxK) * Bt(NxK)^T  (+ epilogue).  4 waves/block stacked on M,
// each wave computes WTM x WTN 16x16 tiles.  All 4 waves share the same
// B tiles, so with TDM=true the 32x(WTN*16) B tile is DMA-staged into LDS
// (double buffered, TENSORcnt-tracked) and read back via ds_load_b128.
// EP: 0 = bf16 out + bias                            (encoder -> h)
//     1 = split: f32 xc_pre | bf16 silu(z)           (W_in)
//     2 = f32 out + bf16 copy of first 64 cols       (W_xproj)
//     3 = softplus(v + bias) -> f32                  (dt projection)
//     4 = bf16 out                                   (W_out -> hs)
//     5 = f32 out + bias                             (decoder / latent)
template <int EP, int WTM, int WTN, bool TDM>
__global__ __launch_bounds__(128) void wmma_gemm_k(
    const bf16_t* __restrict__ A, const bf16_t* __restrict__ Bt,
    const float* __restrict__ bias, void* __restrict__ out0,
    void* __restrict__ out1, int M, int N, int K, int lda, int ldb) {
  const int lane = threadIdx.x & 31;
  const int wave = threadIdx.x >> 5;
  const int m0w  = blockIdx.y * (4 * WTM * 16) + wave * (WTM * 16);
  const int n0w  = blockIdx.x * (WTN * 16);

  f32x8 acc[WTM][WTN];
#pragma unroll
  for (int i = 0; i < WTM; ++i)
#pragma unroll
    for (int j = 0; j < WTN; ++j)
#pragma unroll
      for (int r = 0; r < 8; ++r) acc[i][j][r] = 0.0f;

  if constexpr (TDM) {
    __shared__ bf16_t btile[2][WTN * 16 * 32];
    const uint64_t wbase =
        (uint64_t)(uintptr_t)(Bt + (size_t)n0w * ldb);  // tile row n0w, k=0
    const unsigned int lds0 = (unsigned int)(uintptr_t)(&btile[0][0]);
    const unsigned int lds1 = (unsigned int)(uintptr_t)(&btile[1][0]);
    if (wave == 0) {
      tdm_load_2d(wbase, lds0, (unsigned int)K, (unsigned int)(N - n0w),
                  32u, (unsigned int)(WTN * 16), (uint64_t)ldb);
    }
    int cur = 0;
    for (int k0 = 0; k0 < K; k0 += 32, cur ^= 1) {
      __builtin_amdgcn_s_wait_tensorcnt(0);   // wave0: tile k0 landed in LDS
      __syncthreads();                         // publish to all waves
      if (wave == 0 && (k0 + 32) < K) {        // DMA next tile while computing
        tdm_load_2d(wbase + (uint64_t)(k0 + 32) * 2,
                    cur ? lds0 : lds1, (unsigned int)(K - k0 - 32),
                    (unsigned int)(N - n0w), 32u, (unsigned int)(WTN * 16),
                    (uint64_t)ldb);
      }
      bf16x16 af[WTM];
      bf16x16 bfr[WTN];
#pragma unroll
      for (int i = 0; i < WTM; ++i)
        af[i] = load_a_frag(A, lda, m0w + i * 16, k0, lane);
#pragma unroll
      for (int j = 0; j < WTN; ++j)
        bfr[j] = load_b_frag_lds(&btile[cur][0], j * 16, lane);
#pragma unroll
      for (int i = 0; i < WTM; ++i)
#pragma unroll
        for (int j = 0; j < WTN; ++j)
          acc[i][j] = __builtin_amdgcn_wmma_f32_16x16x32_bf16(
              false, af[i], false, bfr[j], (short)0, acc[i][j], false, false);
    }
  } else {
    for (int k0 = 0; k0 < K; k0 += 32) {
      bf16x16 af[WTM];
      bf16x16 bfr[WTN];
#pragma unroll
      for (int i = 0; i < WTM; ++i)
        af[i] = load_a_frag(A, lda, m0w + i * 16, k0, lane);
#pragma unroll
      for (int j = 0; j < WTN; ++j)
        bfr[j] = load_b_frag(Bt, ldb, n0w + j * 16, k0, lane);
#pragma unroll
      for (int i = 0; i < WTM; ++i)
#pragma unroll
        for (int j = 0; j < WTN; ++j)
          acc[i][j] = __builtin_amdgcn_wmma_f32_16x16x32_bf16(
              false, af[i], false, bfr[j], (short)0, acc[i][j], false, false);
    }
  }

  const int nloc = lane & 15;
  const int mhi  = (lane >> 4) << 3;
#pragma unroll
  for (int i = 0; i < WTM; ++i) {
#pragma unroll
    for (int j = 0; j < WTN; ++j) {
      const int n  = n0w + j * 16 + nloc;
      const int mb = m0w + i * 16 + mhi;
      float bv = 0.0f;
      if constexpr (EP == 0 || EP == 3 || EP == 5) bv = bias[n];
#pragma unroll
      for (int r = 0; r < 8; ++r) {
        const int m = mb + r;
        float v = acc[i][j][r];
        if constexpr (EP == 0) {
          ((bf16_t*)out0)[(size_t)m * N + n] = (bf16_t)(v + bv);
        } else if constexpr (EP == 1) {
          const int Nh = N >> 1;
          if (n < Nh)
            ((float*)out0)[(size_t)m * Nh + n] = v;
          else
            ((bf16_t*)out1)[(size_t)m * Nh + (n - Nh)] = (bf16_t)silu_f(v);
        } else if constexpr (EP == 2) {
          ((float*)out0)[(size_t)m * N + n] = v;
          if (n < DTR_) ((bf16_t*)out1)[(size_t)m * DTR_ + n] = (bf16_t)v;
        } else if constexpr (EP == 3) {
          float s  = v + bv;
          float sp = (s > 20.0f) ? s : log1pf(__expf(s));
          ((float*)out0)[(size_t)m * N + n] = sp;
        } else if constexpr (EP == 4) {
          ((bf16_t*)out0)[(size_t)m * N + n] = (bf16_t)v;
        } else {
          ((float*)out0)[(size_t)m * N + n] = v + bv;
        }
      }
    }
  }
}

// ------------------------------------------------ fp32 -> bf16 convert
__global__ void cvt_bf16_k(const float* __restrict__ s, bf16_t* __restrict__ d, int n) {
  int i = blockIdx.x * blockDim.x + threadIdx.x;
  if (i < n) d[i] = (bf16_t)s[i];
}

// ------------------------------------------------ depthwise causal conv + silu
// out[b,t,d] = silu( sum_{h=0..3} x[b,t+h-3,d]*w[d,h] + cb[d] ), zero-padded.
__global__ __launch_bounds__(256) void conv_silu_k(
    const float* __restrict__ xcpre, const float* __restrict__ cw,
    const float* __restrict__ cb, float* __restrict__ xcf,
    bf16_t* __restrict__ xcb) {
  const int d = blockIdx.x * 256 + threadIdx.x;
  const int t = blockIdx.y;
  const int b = blockIdx.z;
  const float w0 = cw[d * 4 + 0], w1 = cw[d * 4 + 1];
  const float w2 = cw[d * 4 + 2], w3 = cw[d * 4 + 3];
  const size_t idx = ((size_t)b * LL + t) * DI_ + d;
  float acc = cb[d];
  if (t >= 3) acc += xcpre[idx - 3 * (size_t)DI_] * w0;
  if (t >= 2) acc += xcpre[idx - 2 * (size_t)DI_] * w1;
  if (t >= 1) acc += xcpre[idx - 1 * (size_t)DI_] * w2;
  acc += xcpre[idx] * w3;
  const float s = silu_f(acc);
  xcf[idx] = s;
  xcb[idx] = (bf16_t)s;
}

// ------------------------------------------------ selective scan (fused gating)
// One lane per (b,d) channel; 16-state recurrence in registers;
// B_t/C_t staged in LDS; writes g = (y + D*u) * silu(z) as bf16.
__global__ __launch_bounds__(256) void scan_k(
    const float* __restrict__ dt, const float* __restrict__ u,
    const float* __restrict__ dbc, const float* __restrict__ A_log,
    const float* __restrict__ Dskip, const bf16_t* __restrict__ sz,
    bf16_t* __restrict__ g) {
  const int d = blockIdx.x * 256 + threadIdx.x;
  const int b = blockIdx.y;

  float Arow[DS_];
#pragma unroll
  for (int n = 0; n < DS_; ++n) Arow[n] = -__expf(A_log[(size_t)d * DS_ + n]);
  float st[DS_];
#pragma unroll
  for (int n = 0; n < DS_; ++n) st[n] = 0.0f;
  const float Dd = Dskip[d];

  __shared__ float sB[DS_];
  __shared__ float sC[DS_];

  for (int t = 0; t < LL; ++t) {
    const float* bc = dbc + ((size_t)b * LL + t) * 96;
    if (threadIdx.x < 16)
      sB[threadIdx.x] = bc[64 + threadIdx.x];
    else if (threadIdx.x < 32)
      sC[threadIdx.x - 16] = bc[64 + threadIdx.x];
    __syncthreads();
    const size_t idx = ((size_t)b * LL + t) * DI_ + d;
    const float dtv = dt[idx];
    const float uv  = u[idx];
    const float du  = dtv * uv;
    float y = 0.0f;
#pragma unroll
    for (int n = 0; n < DS_; ++n) {
      st[n] = st[n] * __expf(dtv * Arow[n]) + du * sB[n];
      y += st[n] * sC[n];
    }
    const float gv = (y + Dd * uv) * (float)sz[idx];
    g[idx] = (bf16_t)gv;
    __syncthreads();
  }
}

// ---------------------------------------------------------------- launcher
extern "C" void kernel_launch(void* const* d_in, const int* in_sizes, int n_in,
                              void* d_out, int out_size, void* d_ws, size_t ws_size,
                              hipStream_t stream) {
  (void)in_sizes; (void)n_in; (void)out_size; (void)ws_size;
  const float* x        = (const float*)d_in[0];
  const float* W_enc    = (const float*)d_in[1];
  const float* b_enc    = (const float*)d_in[2];
  const float* W_in     = (const float*)d_in[3];
  const float* conv_w   = (const float*)d_in[4];
  const float* conv_b   = (const float*)d_in[5];
  const float* W_xproj  = (const float*)d_in[6];
  const float* W_dtproj = (const float*)d_in[7];
  const float* dt_bias  = (const float*)d_in[8];
  const float* A_log    = (const float*)d_in[9];
  const float* D_skip   = (const float*)d_in[10];
  const float* W_out    = (const float*)d_in[11];
  const float* W_dec    = (const float*)d_in[12];
  const float* b_dec    = (const float*)d_in[13];
  const float* W_lat    = (const float*)d_in[14];
  const float* b_lat    = (const float*)d_in[15];

  // ---- workspace carve (256B aligned) ----
  char* wsp = (char*)d_ws;
  size_t off = 0;
  auto carve = [&](size_t bytes) -> char* {
    char* p = wsp + off;
    off += (bytes + 255) & ~(size_t)255;
    return p;
  };
  bf16_t* x_bf    = (bf16_t*)carve((size_t)MROWS * OBS_ * 2);
  bf16_t* Wenc_bf = (bf16_t*)carve((size_t)DM_ * OBS_ * 2);
  bf16_t* Win_bf  = (bf16_t*)carve((size_t)2 * DI_ * DM_ * 2);
  bf16_t* Wxp_bf  = (bf16_t*)carve((size_t)(DTR_ + 2 * DS_) * DI_ * 2);
  bf16_t* Wdt_bf  = (bf16_t*)carve((size_t)DI_ * DTR_ * 2);
  bf16_t* Wout_bf = (bf16_t*)carve((size_t)DM_ * DI_ * 2);
  bf16_t* Wdec_bf = (bf16_t*)carve((size_t)OBS_ * DM_ * 2);
  bf16_t* Wlat_bf = (bf16_t*)carve((size_t)LAT_ * DM_ * 2);
  bf16_t* h_bf    = (bf16_t*)carve((size_t)MROWS * DM_ * 2);
  float*  xcpre_f = (float*) carve((size_t)MROWS * DI_ * 4);
  bf16_t* sz_bf   = (bf16_t*)carve((size_t)MROWS * DI_ * 2);
  float*  xc_f    = (float*) carve((size_t)MROWS * DI_ * 4);
  bf16_t* xc_bf   = (bf16_t*)carve((size_t)MROWS * DI_ * 2);
  float*  dbc_f   = (float*) carve((size_t)MROWS * 96 * 4);
  bf16_t* dtr_bf  = (bf16_t*)carve((size_t)MROWS * DTR_ * 2);
  float*  dt_f    = (float*) carve((size_t)MROWS * DI_ * 4);
  // lifetime-based aliases:
  bf16_t* g_bf  = (bf16_t*)xcpre_f;  // xc_pre dead after conv
  bf16_t* hs_bf = h_bf;              // h dead after W_in GEMM

  auto cvt = [&](const float* src, bf16_t* dst, size_t n) {
    int blocks = (int)((n + 255) / 256);
    cvt_bf16_k<<<dim3(blocks), dim3(256), 0, stream>>>(src, dst, (int)n);
  };
  cvt(x,        x_bf,    (size_t)MROWS * OBS_);
  cvt(W_enc,    Wenc_bf, (size_t)DM_ * OBS_);
  cvt(W_in,     Win_bf,  (size_t)2 * DI_ * DM_);
  cvt(W_xproj,  Wxp_bf,  (size_t)(DTR_ + 2 * DS_) * DI_);
  cvt(W_dtproj, Wdt_bf,  (size_t)DI_ * DTR_);
  cvt(W_out,    Wout_bf, (size_t)DM_ * DI_);
  cvt(W_dec,    Wdec_bf, (size_t)OBS_ * DM_);
  cvt(W_lat,    Wlat_bf, (size_t)LAT_ * DM_);

  // 1) h = x @ W_enc^T + b_enc                    (M=4096,N=1024,K=128)
  wmma_gemm_k<0, 2, 2, true><<<dim3(DM_ / 32, MROWS / 128), 128, 0, stream>>>(
      x_bf, Wenc_bf, b_enc, h_bf, nullptr, MROWS, DM_, OBS_, OBS_, OBS_);

  // 2) xz = h @ W_in^T -> xc_pre f32 | silu(z) bf16  (M=4096,N=4096,K=1024)
  wmma_gemm_k<1, 2, 2, true><<<dim3((2 * DI_) / 32, MROWS / 128), 128, 0, stream>>>(
      h_bf, Win_bf, nullptr, xcpre_f, sz_bf, MROWS, 2 * DI_, DM_, DM_, DM_);

  // 3) depthwise causal conv + bias + silu
  conv_silu_k<<<dim3(DI_ / 256, LL, BB), 256, 0, stream>>>(
      xcpre_f, conv_w, conv_b, xc_f, xc_bf);

  // 4) dbc = xc @ W_xproj^T                       (M=4096,N=96,K=2048)
  wmma_gemm_k<2, 2, 2, true><<<dim3(96 / 32, MROWS / 128), 128, 0, stream>>>(
      xc_bf, Wxp_bf, nullptr, dbc_f, dtr_bf, MROWS, 96, DI_, DI_, DI_);

  // 5) dt = softplus(dbc[:, :64] @ W_dtproj^T + dt_bias)  (M=4096,N=2048,K=64)
  wmma_gemm_k<3, 2, 2, true><<<dim3(DI_ / 32, MROWS / 128), 128, 0, stream>>>(
      dtr_bf, Wdt_bf, dt_bias, dt_f, nullptr, MROWS, DI_, DTR_, DTR_, DTR_);

  // 6) selective scan + fused (y + D*u)*silu(z) gating -> g bf16
  scan_k<<<dim3(DI_ / 256, BB), 256, 0, stream>>>(
      dt_f, xc_f, dbc_f, A_log, D_skip, sz_bf, g_bf);

  // 7) hs = g @ W_out^T                           (M=4096,N=1024,K=2048)
  wmma_gemm_k<4, 2, 2, true><<<dim3(DM_ / 32, MROWS / 128), 128, 0, stream>>>(
      g_bf, Wout_bf, nullptr, hs_bf, nullptr, MROWS, DM_, DI_, DI_, DI_);

  // 8) recon = hs @ W_dec^T + b_dec -> d_out[0:524288]   (N=128,K=1024)
  wmma_gemm_k<5, 2, 2, true><<<dim3(OBS_ / 32, MROWS / 128), 128, 0, stream>>>(
      hs_bf, Wdec_bf, b_dec, (float*)d_out, nullptr, MROWS, OBS_, DM_, DM_, DM_);

  // 9) lat = hs @ W_lat^T + b_lat -> d_out[524288:]      (N=16,K=1024)
  wmma_gemm_k<5, 2, 1, true><<<dim3(LAT_ / 16, MROWS / 128), 128, 0, stream>>>(
      hs_bf, Wlat_bf, b_lat, (float*)d_out + (size_t)MROWS * OBS_, nullptr,
      MROWS, LAT_, DM_, DM_, DM_);
}
